// SparseMoE_24532853195084
// MI455X (gfx1250) — matrix operand order repaired and checked by reading, not compiled, not run on hardware
//
#include <hip/hip_runtime.h>
#include <hip/hip_bf16.h>

// Problem constants (match reference)
#define B_ 8
#define S_ 2048
#define D_ 1024
#define H_ 4096
#define E_ 8
#define K_ 2

typedef __attribute__((ext_vector_type(16))) __bf16 bf16x16;
typedef __attribute__((ext_vector_type(2)))  __bf16 bf16x2;
typedef __attribute__((ext_vector_type(8)))  float  f32x8;

// Idiomatic pair-convert: lowers to a single v_cvt_pk_bf16_f32.
static __device__ __forceinline__ unsigned pack_bf16x2(float lo, float hi) {
  bf16x2 t;
  t[0] = (__bf16)lo;
  t[1] = (__bf16)hi;
  return __builtin_bit_cast(unsigned, t);
}

static __device__ __forceinline__ float silu_f(float v) {
  return v / (1.0f + __expf(-v));
}

// ---------------------------------------------------------------------------
// 1) routing_input = mean(x, axis=S)   grid (D_/256, B_), block 256
__global__ __launch_bounds__(256)
void mean_kernel(const float* __restrict__ x, float* __restrict__ meanb) {
  const int b = blockIdx.y;
  const int d = blockIdx.x * 256 + threadIdx.x;
  const float* p = x + (size_t)b * S_ * D_ + d;
  float s = 0.0f;
  for (int i = 0; i < S_; ++i) s += p[(size_t)i * D_];
  meanb[b * D_ + d] = s * (1.0f / (float)S_);
}

// ---------------------------------------------------------------------------
// 2) gate MLP + top-2 + softmax. One block per batch element.
__global__ __launch_bounds__(256)
void gate_kernel(const float* __restrict__ meanb, const float* __restrict__ Wg1,
                 const float* __restrict__ bg1, const float* __restrict__ Wg2,
                 const float* __restrict__ bg2,
                 float* __restrict__ topw, int* __restrict__ topidx,
                 float* __restrict__ outw, int* __restrict__ outidx) {
  __shared__ float smean[D_];
  __shared__ float sgh[D_];
  __shared__ float spart[256 * E_];
  __shared__ float slog[E_];

  const int b = blockIdx.x;
  const int tid = threadIdx.x;

  for (int d = tid; d < D_; d += 256) smean[d] = meanb[b * D_ + d];
  __syncthreads();

  for (int d = tid; d < D_; d += 256) {
    float s = bg1[d];
    for (int i = 0; i < D_; ++i) s += smean[i] * Wg1[(size_t)i * D_ + d];
    sgh[d] = silu_f(s);
  }
  __syncthreads();

  float p[E_];
#pragma unroll
  for (int e = 0; e < E_; ++e) p[e] = 0.0f;
  for (int d = tid; d < D_; d += 256) {
    float g = sgh[d];
#pragma unroll
    for (int e = 0; e < E_; ++e) p[e] += g * Wg2[d * E_ + e];
  }
#pragma unroll
  for (int e = 0; e < E_; ++e) spart[tid * E_ + e] = p[e];
  __syncthreads();

  if (tid < E_) {
    float s = bg2[tid];
    for (int t = 0; t < 256; ++t) s += spart[t * E_ + tid];
    slog[tid] = s;
  }
  __syncthreads();

  if (tid == 0) {
    int i0 = 0; float l0 = slog[0];
#pragma unroll
    for (int e = 1; e < E_; ++e) { if (slog[e] > l0) { l0 = slog[e]; i0 = e; } }
    int i1 = (i0 == 0) ? 1 : 0; float l1 = slog[i1];
#pragma unroll
    for (int e = 0; e < E_; ++e) {
      if (e != i0 && slog[e] > l1) { l1 = slog[e]; i1 = e; }
    }
    float e1 = __expf(l1 - l0);
    float w0 = 1.0f / (1.0f + e1);
    float w1 = e1 / (1.0f + e1);
    topw[b * K_ + 0] = w0; topw[b * K_ + 1] = w1;
    topidx[b * K_ + 0] = i0; topidx[b * K_ + 1] = i1;
    outw[b * K_ + 0] = w0; outw[b * K_ + 1] = w1;
    outidx[b * K_ + 0] = i0; outidx[b * K_ + 1] = i1;
  }
}

// ---------------------------------------------------------------------------
// 3) bf16-WMMA GEMM. fp32 sources converted in the loader (v_cvt_pk_bf16_f32),
//    double-buffered LDS with XOR chunk swizzle (conflict-free ds_load_b128),
//    BMx128 block tile, K-step 32, 8 waves.
//    BM=128: wave tile 32x64 (2x4 WMMAs/step). BM=64: wave tile 32x32 (2x2).
// LDS rows are 32 bf16 = 4 chunks of 16B; chunk lc of row r lives at physical
// chunk lc ^ ((r>>2)&3). Lanes {i,i+4,i+8,i+12} then hit distinct banks.
// MODE 0: C = silu(A@B+bias)    MODE 1: C = s*(A@B+bias)    MODE 2: C += s*(..)
template <int MODE, int BM>
__global__ __launch_bounds__(256)
void moe_gemm(const float* __restrict__ A, const float* __restrict__ Wbase,
              long long wstride, const float* __restrict__ biasBase, int biasStride,
              float* __restrict__ C, int Kdim, int lda, int ldb, int ldc,
              const int* __restrict__ topidx, const float* __restrict__ topw,
              int bk) {
  constexpr int BN  = 128;
  constexpr int WGM = BM / 32;          // waves along M (4 or 2)
  constexpr int WGN = 8 / WGM;          // waves along N (2 or 4)
  constexpr int CR  = 2;                // 16-row subtiles per wave
  constexpr int CC  = (BN / WGN) / 16;  // 16-col subtiles per wave (4 or 2)
  constexpr int AL  = BM / 32;          // float4 loads per thread for A (4 or 2)

  __shared__ __bf16 sA[2][BM * 32];     // [m][k], chunk-swizzled
  __shared__ __bf16 sBT[2][BN * 32];    // [n][k] transposed, chunk-swizzled

  const int tid  = threadIdx.x;
  const int lane = tid & 31;
  const int wave = tid >> 5;
  const int wm   = wave / WGN;
  const int wn   = wave % WGN;
  const int mbase = blockIdx.y * BM;
  const int nbase = blockIdx.x * BN;

  const int e = topidx[bk];
  const float* Bmat = Wbase + (long long)e * wstride;
  const float* bias = biasBase + (long long)e * biasStride;

  f32x8 acc[CR][CC];
#pragma unroll
  for (int r = 0; r < CR; ++r)
#pragma unroll
    for (int c = 0; c < CC; ++c)
      acc[r][c] = (f32x8){0.f, 0.f, 0.f, 0.f, 0.f, 0.f, 0.f, 0.f};

  // A loader: thread owns one row-segment of AL*4 columns.
  const int ar = tid / (256 / BM);
  const int ac = (tid % (256 / BM)) * AL * 4;
  const int aswz = (ar >> 2) & 3;
  // B loader: thread owns a 2(k) x 8(n) sub-block -> packed b32 transposed stores.
  const int rp = tid >> 4;        // k rows rp*2, rp*2+1
  const int cg = tid & 15;        // n cols cg*8 .. +7

  const int lm  = lane & 15;
  const int c0A = (lane < 16) ? 0 : 1;  // logical 16B chunk of A (k 0..7 / 8..15)
  const int c0B = (lane < 16) ? 0 : 1;  // logical 16B chunk of B (k 0..7 / 8..15)

  float4 areg[AL];
  float4 breg[4];

  auto fetch = [&](int kt) {
    const float* ap = A + (long long)(mbase + ar) * lda + kt + ac;
#pragma unroll
    for (int i = 0; i < AL; ++i) areg[i] = *(const float4*)(ap + 4 * i);
    const float* bp = Bmat + (long long)(kt + rp * 2) * ldb + nbase + cg * 8;
    breg[0] = *(const float4*)bp;
    breg[1] = *(const float4*)(bp + 4);
    breg[2] = *(const float4*)(bp + ldb);
    breg[3] = *(const float4*)(bp + ldb + 4);
    if (kt + 32 < Kdim) __builtin_prefetch(bp + (long long)32 * ldb, 0, 0);
  };

  auto stage = [&](int buf) {
    // A: packed 16B stores at swizzled chunk position
#pragma unroll
    for (int i = 0; i < AL; i += 2) {
      uint4 u;
      u.x = pack_bf16x2(areg[i].x, areg[i].y);
      u.y = pack_bf16x2(areg[i].z, areg[i].w);
      u.z = pack_bf16x2(areg[i + 1].x, areg[i + 1].y);
      u.w = pack_bf16x2(areg[i + 1].z, areg[i + 1].w);
      const int lc = ac / 8 + i / 2;
      *(uint4*)&sA[buf][ar * 32 + ((lc ^ aswz) * 8)] = u;
    }
    // B transposed: pack (k,k+1) pairs -> 8 x ds_store_b32 at swizzled chunks
    float lo[8] = {breg[0].x, breg[0].y, breg[0].z, breg[0].w,
                   breg[1].x, breg[1].y, breg[1].z, breg[1].w};
    float hi[8] = {breg[2].x, breg[2].y, breg[2].z, breg[2].w,
                   breg[3].x, breg[3].y, breg[3].z, breg[3].w};
    const int kchunk = rp >> 2;          // 16B chunk along k
    const int koff   = (rp * 2) & 7;     // element within chunk
#pragma unroll
    for (int j = 0; j < 8; ++j) {
      const int n = cg * 8 + j;
      const int sw = (n >> 2) & 3;
      *(unsigned*)&sBT[buf][n * 32 + ((kchunk ^ sw) * 8) + koff] =
          pack_bf16x2(lo[j], hi[j]);
    }
  };

  union Frag { bf16x16 v; uint4 q[2]; };

  auto compute = [&](int buf) {
    Frag fa[CR], fb[CC];
#pragma unroll
    for (int r = 0; r < CR; ++r) {
      const int m = wm * 32 + r * 16 + lm;
      const int sw = (m >> 2) & 3;
      fa[r].q[0] = *(const uint4*)&sA[buf][m * 32 + ((c0A ^ sw) * 8)];
      fa[r].q[1] = *(const uint4*)&sA[buf][m * 32 + (((c0A + 2) ^ sw) * 8)];
    }
#pragma unroll
    for (int c = 0; c < CC; ++c) {
      const int n = wn * (CC * 16) + c * 16 + lm;
      const int sw = (n >> 2) & 3;
      fb[c].q[0] = *(const uint4*)&sBT[buf][n * 32 + ((c0B ^ sw) * 8)];
      fb[c].q[1] = *(const uint4*)&sBT[buf][n * 32 + (((c0B + 2) ^ sw) * 8)];
    }
#pragma unroll
    for (int r = 0; r < CR; ++r)
#pragma unroll
      for (int c = 0; c < CC; ++c)
        acc[r][c] = __builtin_amdgcn_wmma_f32_16x16x32_bf16(
            false, fa[r].v, false, fb[c].v, (short)0, acc[r][c], false, false);
  };

  fetch(0);
  stage(0);
  __syncthreads();

  int buf = 0;
  for (int kt = 0; kt + 32 < Kdim; kt += 32) {   // steady state: no branches
    fetch(kt + 32);
    compute(buf);
    stage(buf ^ 1);
    __syncthreads();
    buf ^= 1;
  }
  compute(buf);                                   // peeled last K-step

  // epilogue: VGPR v -> M = v (lanes 0-15) or v+8 (lanes 16-31); lane -> N
  const float scale = (MODE == 0) ? 1.0f : topw[bk];
  const int moff = (lane < 16) ? 0 : 8;
#pragma unroll
  for (int r = 0; r < CR; ++r) {
#pragma unroll
    for (int c = 0; c < CC; ++c) {
      const int n = nbase + wn * (CC * 16) + c * 16 + lm;
      const float bv = bias[n];
      const int m0 = mbase + wm * 32 + r * 16 + moff;
#pragma unroll
      for (int v = 0; v < 8; ++v) {
        float val = acc[r][c][v] + bv;
        const long long idx = (long long)(m0 + v) * ldc + n;
        if (MODE == 0)      C[idx] = silu_f(val);
        else if (MODE == 1) C[idx] = val * scale;
        else                C[idx] += val * scale;
      }
    }
  }
}

// ---------------------------------------------------------------------------
extern "C" void kernel_launch(void* const* d_in, const int* in_sizes, int n_in,
                              void* d_out, int out_size, void* d_ws, size_t ws_size,
                              hipStream_t stream) {
  const float* x   = (const float*)d_in[0];
  const float* Wg1 = (const float*)d_in[1];
  const float* bg1 = (const float*)d_in[2];
  const float* Wg2 = (const float*)d_in[3];
  const float* bg2 = (const float*)d_in[4];
  const float* W1  = (const float*)d_in[5];
  const float* b1  = (const float*)d_in[6];
  const float* W2  = (const float*)d_in[7];
  const float* b2  = (const float*)d_in[8];
  float* out = (float*)d_out;

  // workspace: h[S*H] | mean[B*D] | topw[B*K] | topidx[B*K](int)
  float* h      = (float*)d_ws;
  float* meanb  = h + (size_t)S_ * H_;
  float* topw   = meanb + (size_t)B_ * D_;
  int*   topidx = (int*)(topw + B_ * K_);

  // d_out: [B,S,D] fp32, then topk_w [B,K] fp32, then topk_idx [B,K] int32 bits
  float* outw   = out + (size_t)B_ * S_ * D_;
  int*   outidx = (int*)(outw + B_ * K_);

  mean_kernel<<<dim3(D_ / 256, B_), 256, 0, stream>>>(x, meanb);
  gate_kernel<<<B_, 256, 0, stream>>>(meanb, Wg1, bg1, Wg2, bg2,
                                      topw, topidx, outw, outidx);

  for (int b = 0; b < B_; ++b) {
    for (int k = 0; k < K_; ++k) {
      const int bk = b * K_ + k;
      // h = silu(x[b] @ W1[e] + b1[e])   [S,H]  : 128x128 tiles
      moe_gemm<0, 128><<<dim3(H_ / 128, S_ / 128), 256, 0, stream>>>(
          x + (size_t)b * S_ * D_, W1, (long long)D_ * H_, b1, H_,
          h, D_, D_, H_, H_, topidx, topw, bk);
      // out[b] (+)= w * (h @ W2[e] + b2[e])   [S,D] : 64x128 tiles
      if (k == 0) {
        moe_gemm<1, 64><<<dim3(D_ / 128, S_ / 64), 256, 0, stream>>>(
            h, W2, (long long)H_ * D_, b2, D_,
            out + (size_t)b * S_ * D_, H_, H_, D_, D_, topidx, topw, bk);
      } else {
        moe_gemm<2, 64><<<dim3(D_ / 128, S_ / 64), 256, 0, stream>>>(
            h, W2, (long long)H_ * D_, b2, D_,
            out + (size_t)b * S_ * D_, H_, H_, D_, D_, topidx, topw, bk);
      }
    }
  }
}
